// SwinJSCC_Encoder_68307159875696
// MI455X (gfx1250) — compile-verified, matrix operand order checked by reading, not run on hardware
//
#include <hip/hip_runtime.h>
#include <math.h>

// ---------------------------------------------------------------------------
// CDNA5 (gfx1250) Swin encoder. All GEMMs via v_wmma_f32_16x16x32_bf16.
// wave32; 128-thread blocks = 4 waves for WMMA kernels.
// ---------------------------------------------------------------------------

typedef __attribute__((ext_vector_type(16))) __bf16 bf16x16;
typedef __attribute__((ext_vector_type(2)))  __bf16 bf16x2;
typedef __attribute__((ext_vector_type(8)))  float  f32x8;

#define LDT 34   // LDS tile stride for 32-wide bf16 tiles (keeps 4B alignment, kills bank conflicts)

// K-index pattern for 16-bit A/B fragments (ISA 7.12.2, 16-bit A 16x32):
// VGPR pair v (0..7), half hi (lane>>4): base K = (v&3)*2 + hi*8 + (v>>2)*16
__device__ __forceinline__ int frag_k(int v, int hi) {
    return ((v & 3) * 2) + hi * 8 + ((v >> 2) * 16);
}

// ---------------------------------------------------------------------------
// Generic GEMM: out[M,N] = A[M,K] (fp32) x W[K,N] (fp32) + bias, epilogue MODE
// MODE 0: none   MODE 1: exact GELU   MODE 2: += res[M,N]
// Block: 128 thr (4 waves, 2x2), tile 64x64, K-step 32.
// ---------------------------------------------------------------------------
template <int MODE>
__global__ __launch_bounds__(128) void gemm_wmma(
    const float* __restrict__ A, const float* __restrict__ W,
    const float* __restrict__ bias, const float* __restrict__ res,
    float* __restrict__ out, int M, int N, int K)
{
    __shared__ __bf16 As[64][LDT];
    __shared__ __bf16 Bs[64][LDT];   // stored transposed: [n][k]

    const int t    = threadIdx.x;
    const int lane = t & 31;
    const int wave = t >> 5;
    const int wm   = wave >> 1, wn = wave & 1;
    const int hi   = lane >> 4, mr = lane & 15;
    const int m0   = blockIdx.y * 64;
    const int n0   = blockIdx.x * 64;

    f32x8 acc[2][2] = {};

    for (int k0 = 0; k0 < K; k0 += 32) {
        // Stage A tile 64x32 (fp32 -> bf16), coalesced on k
        for (int i = t; i < 64 * 32; i += 128) {
            int r = i >> 5, c = i & 31;
            int gm = m0 + r;
            float v = (gm < M) ? A[(size_t)gm * K + (k0 + c)] : 0.0f;
            As[r][c] = (__bf16)v;
        }
        // Stage B tile 32x64 transposed -> [n][k], coalesced on n
        for (int i = t; i < 64 * 32; i += 128) {
            int kr = i >> 6, nc = i & 63;
            int gn = n0 + nc;
            float v = (gn < N) ? W[(size_t)(k0 + kr) * N + gn] : 0.0f;
            Bs[nc][kr] = (__bf16)v;
        }
        __syncthreads();

        bf16x16 afr[2], bfr[2];
        for (int mt = 0; mt < 2; ++mt) {
            int row = 32 * wm + 16 * mt + mr;
            for (int v = 0; v < 8; ++v) {
                bf16x2 p = *(const bf16x2*)&As[row][frag_k(v, hi)];
                afr[mt][2 * v] = p.x; afr[mt][2 * v + 1] = p.y;
            }
        }
        for (int nt = 0; nt < 2; ++nt) {
            int col = 32 * wn + 16 * nt + mr;
            for (int v = 0; v < 8; ++v) {
                bf16x2 p = *(const bf16x2*)&Bs[col][frag_k(v, hi)];
                bfr[nt][2 * v] = p.x; bfr[nt][2 * v + 1] = p.y;
            }
        }
        for (int mt = 0; mt < 2; ++mt)
            for (int nt = 0; nt < 2; ++nt)
                acc[mt][nt] = __builtin_amdgcn_wmma_f32_16x16x32_bf16(
                    false, afr[mt], false, bfr[nt], (short)0, acc[mt][nt], false, false);
        __syncthreads();
    }

    // Epilogue: C/D layout -> row = r + 8*hi, col = mr (per 16x16 tile)
    const int mbase = m0 + 32 * wm + 8 * hi;
    for (int mt = 0; mt < 2; ++mt) {
        for (int nt = 0; nt < 2; ++nt) {
            int gn = n0 + 32 * wn + 16 * nt + mr;
            if (gn >= N) continue;
            float bv = bias ? bias[gn] : 0.0f;
            for (int r = 0; r < 8; ++r) {
                int gm = mbase + 16 * mt + r;
                if (gm >= M) continue;
                float v = acc[mt][nt][r] + bv;
                if (MODE == 1) v = 0.5f * v * (1.0f + erff(v * 0.70710678118654752f));
                else if (MODE == 2) v += res[(size_t)gm * N + gn];
                out[(size_t)gm * N + gn] = v;
            }
        }
    }
}

// ---------------------------------------------------------------------------
// Fused window attention. One workgroup per (global window, head).
// qkv: [numWin*64, 3C] fp32 (col layout: s*C + h*32 + d). O -> [numWin*64, C].
// ---------------------------------------------------------------------------
__device__ __forceinline__ int shift_region(int r, int H) {
    return (r < H - 8) ? 0 : ((r < H - 4) ? 1 : 2);
}

__global__ __launch_bounds__(128) void win_attn(
    const float* __restrict__ qkv, float* __restrict__ outO,
    const float* __restrict__ rpb, int C, int nh, int nWs, int Hres, int shift)
{
    __shared__ __bf16 Qs[64][LDT];   // [token][dim], pre-scaled
    __shared__ __bf16 Ks[64][LDT];   // [token][dim]  == B-frag source [n][k] for S
    __shared__ __bf16 Vt[32][66];    // [dim][token]  == B-frag source [n][k] for O
    __shared__ float  Sf[64][65];    // scores / probs (fp32)
    __shared__ __bf16 Pb[64][66];    // probs bf16 (A-matrix for O)

    const int t    = threadIdx.x;
    const int lane = t & 31;
    const int wave = t >> 5;
    const int hi   = lane >> 4, mr = lane & 15;
    const int win  = blockIdx.x;
    const int h    = blockIdx.y;
    const int C3   = 3 * C;
    const size_t rowbase = (size_t)win * 64;
    const float scale = 0.17677669529663687f;  // 1/sqrt(32)

    for (int i = t; i < 64 * 32; i += 128) {
        int tok = i >> 5, d = i & 31;
        size_t ro = (rowbase + tok) * (size_t)C3;
        int col = h * 32 + d;
        Qs[tok][d] = (__bf16)(qkv[ro + col] * scale);
        Ks[tok][d] = (__bf16)(qkv[ro + C + col]);
        Vt[d][tok] = (__bf16)(qkv[ro + 2 * C + col]);
    }
    __syncthreads();

    // ---- S = Q * K^T : wave handles rows 16*wave .. +15 (K-dim = 32) ----
    bf16x16 aq;
    {
        int row = 16 * wave + mr;
        for (int v = 0; v < 8; ++v) {
            bf16x2 p = *(const bf16x2*)&Qs[row][frag_k(v, hi)];
            aq[2 * v] = p.x; aq[2 * v + 1] = p.y;
        }
    }
    f32x8 sacc[4] = {};
    for (int nt = 0; nt < 4; ++nt) {
        bf16x16 bk;
        int col = 16 * nt + mr;
        for (int v = 0; v < 8; ++v) {
            bf16x2 p = *(const bf16x2*)&Ks[col][frag_k(v, hi)];
            bk[2 * v] = p.x; bk[2 * v + 1] = p.y;
        }
        sacc[nt] = __builtin_amdgcn_wmma_f32_16x16x32_bf16(
            false, aq, false, bk, (short)0, sacc[nt], false, false);
    }

    // ---- relative-position bias + shift mask, scatter to Sf ----
    const int nWimg = nWs * nWs;
    const int wimg  = win % nWimg;
    const int wr = wimg / nWs, wc = wimg % nWs;
    for (int nt = 0; nt < 4; ++nt) {
        int n  = 16 * nt + mr;
        int ni = n >> 3, nj = n & 7;
        for (int r = 0; r < 8; ++r) {
            int m  = 16 * wave + 8 * hi + r;
            int mi = m >> 3, mj = m & 7;
            int idx = (mi - ni + 7) * 15 + (mj - nj + 7);
            float v = sacc[nt][r] + rpb[idx * nh + h];
            if (shift) {
                int lm = shift_region(wr * 8 + mi, Hres) * 3 + shift_region(wc * 8 + mj, Hres);
                int ln = shift_region(wr * 8 + ni, Hres) * 3 + shift_region(wc * 8 + nj, Hres);
                if (lm != ln) v -= 100.0f;
            }
            Sf[m][n] = v;
        }
    }
    __syncthreads();

    // ---- softmax over 64 cols, one thread per row ----
    if (t < 64) {
        float mx = -1e30f;
        for (int n = 0; n < 64; ++n) mx = fmaxf(mx, Sf[t][n]);
        float s = 0.0f;
        for (int n = 0; n < 64; ++n) { float e = __expf(Sf[t][n] - mx); Sf[t][n] = e; s += e; }
        float inv = 1.0f / s;
        for (int n = 0; n < 64; ++n) Pb[t][n] = (__bf16)(Sf[t][n] * inv);
    }
    __syncthreads();

    // ---- O = P * V : K-dim 64 (2 steps), N = 32 dims (2 tiles) ----
    f32x8 oacc[2] = {};
    for (int ks = 0; ks < 2; ++ks) {
        bf16x16 ap;
        int row = 16 * wave + mr;
        for (int v = 0; v < 8; ++v) {
            bf16x2 p = *(const bf16x2*)&Pb[row][32 * ks + frag_k(v, hi)];
            ap[2 * v] = p.x; ap[2 * v + 1] = p.y;
        }
        for (int nt = 0; nt < 2; ++nt) {
            bf16x16 bv;
            int col = 16 * nt + mr;
            for (int v = 0; v < 8; ++v) {
                bf16x2 p = *(const bf16x2*)&Vt[col][32 * ks + frag_k(v, hi)];
                bv[2 * v] = p.x; bv[2 * v + 1] = p.y;
            }
            oacc[nt] = __builtin_amdgcn_wmma_f32_16x16x32_bf16(
                false, ap, false, bv, (short)0, oacc[nt], false, false);
        }
    }
    for (int nt = 0; nt < 2; ++nt) {
        int d = 16 * nt + mr;
        for (int r = 0; r < 8; ++r) {
            int m = 16 * wave + 8 * hi + r;
            outO[(rowbase + m) * (size_t)C + h * 32 + d] = oacc[nt][r];
        }
    }
}

// ---------------------------------------------------------------------------
// LayerNorm: one 128-thread block per row.
// ---------------------------------------------------------------------------
__global__ __launch_bounds__(128) void ln_kernel(
    const float* __restrict__ in, float* __restrict__ out,
    const float* __restrict__ w, const float* __restrict__ b, int C)
{
    const int t = threadIdx.x;
    const size_t row = blockIdx.x;
    const float* p = in + row * (size_t)C;
    float s = 0.0f, s2 = 0.0f;
    for (int c = t; c < C; c += 128) { float v = p[c]; s += v; s2 += v * v; }
    __shared__ float rs[128], rq[128];
    rs[t] = s; rq[t] = s2;
    __syncthreads();
    for (int o = 64; o > 0; o >>= 1) {
        if (t < o) { rs[t] += rs[t + o]; rq[t] += rq[t + o]; }
        __syncthreads();
    }
    float mean = rs[0] / (float)C;
    float var  = rq[0] / (float)C - mean * mean;
    float inv  = rsqrtf(var + 1e-5f);
    for (int c = t; c < C; c += 128)
        out[row * (size_t)C + c] = (p[c] - mean) * inv * w[c] + b[c];
}

// ---------------------------------------------------------------------------
// PatchEmbed conv 2x2 s2: x[B,3,256,256] -> out[B,128*128,128] (+bias)
// ---------------------------------------------------------------------------
__global__ void patch_embed(const float* __restrict__ x, const float* __restrict__ w,
                            const float* __restrict__ bias, float* __restrict__ out, int Bq)
{
    size_t idx = (size_t)blockIdx.x * blockDim.x + threadIdx.x;
    size_t total = (size_t)Bq * 128 * 128 * 128;
    if (idx >= total) return;
    int oc  = idx & 127;
    int pos = (int)((idx >> 7) & (128 * 128 - 1));
    int b   = (int)(idx >> 21);
    int oy = pos >> 7, ox = pos & 127;
    float s = bias[oc];
    for (int ic = 0; ic < 3; ++ic)
        for (int ky = 0; ky < 2; ++ky)
            for (int kx = 0; kx < 2; ++kx)
                s += x[(((size_t)b * 3 + ic) * 256 + (2 * oy + ky)) * 256 + (2 * ox + kx)]
                   * w[((oc * 3 + ic) * 2 + ky) * 2 + kx];
    out[idx] = s;
}

// ---------------------------------------------------------------------------
// Window partition with cyclic shift (roll by -shift folded into gather).
// in: [B, H, H, C] -> out: window-ordered rows [B*nW*64, C]
// ---------------------------------------------------------------------------
__global__ void win_part(const float* __restrict__ in, float* __restrict__ out,
                         int Bq, int Hres, int C, int shift)
{
    size_t idx = (size_t)blockIdx.x * blockDim.x + threadIdx.x;
    size_t total = (size_t)Bq * Hres * Hres * C;
    if (idx >= total) return;
    int c = (int)(idx % C);
    size_t rrow = idx / C;
    int tok = (int)(rrow & 63);
    size_t wg = rrow >> 6;
    int nWs = Hres >> 3;
    int nWimg = nWs * nWs;
    int wimg = (int)(wg % nWimg);
    int b    = (int)(wg / nWimg);
    int wr = wimg / nWs, wc = wimg % nWs;
    int y = wr * 8 + (tok >> 3);
    int x = wc * 8 + (tok & 7);
    if (shift) { y = (y + shift) % Hres; x = (x + shift) % Hres; }
    out[idx] = in[(((size_t)b * Hres + y) * Hres + x) * C + c];
}

// ---------------------------------------------------------------------------
// Window reverse + unshift + residual add: xio[b,y,x,c] += po[window row]
// ---------------------------------------------------------------------------
__global__ void win_rev_add(const float* __restrict__ po, float* __restrict__ xio,
                            int Bq, int Hres, int C, int shift)
{
    size_t idx = (size_t)blockIdx.x * blockDim.x + threadIdx.x;
    size_t total = (size_t)Bq * Hres * Hres * C;
    if (idx >= total) return;
    int c = (int)(idx % C);
    size_t rrow = idx / C;
    int pix = (int)(rrow % (Hres * Hres));
    int b   = (int)(rrow / (Hres * Hres));
    int y = pix / Hres, x = pix % Hres;
    int ry = y, rx = x;
    if (shift) { ry = (y - shift + Hres) % Hres; rx = (x - shift + Hres) % Hres; }
    int nWs = Hres >> 3;
    size_t wrow = (((size_t)b * nWs * nWs) + (size_t)(ry >> 3) * nWs + (rx >> 3)) * 64
                + (size_t)(ry & 7) * 8 + (rx & 7);
    xio[idx] += po[wrow * (size_t)C + c];
}

// ---------------------------------------------------------------------------
// Patch merge gather: [B,Hin,Hin,C] -> [B,(Hin/2)^2, 4C], quadrant order
// (0,0),(1,0),(0,1),(1,1)
// ---------------------------------------------------------------------------
__global__ void merge_gather(const float* __restrict__ in, float* __restrict__ out,
                             int Bq, int Hin, int C)
{
    int Ho = Hin >> 1;
    size_t idx = (size_t)blockIdx.x * blockDim.x + threadIdx.x;
    size_t total = (size_t)Bq * Ho * Ho * 4 * C;
    if (idx >= total) return;
    int cc = (int)(idx % (4 * C));
    size_t orow = idx / (4 * C);
    int q = cc / C, c = cc % C;
    int pix = (int)(orow % (Ho * Ho));
    int b   = (int)(orow / (Ho * Ho));
    int oy = pix / Ho, ox = pix % Ho;
    int dy = (q == 1 || q == 3) ? 1 : 0;
    int dx = (q >= 2) ? 1 : 0;
    out[idx] = in[(((size_t)b * Hin + (2 * oy + dy)) * Hin + (2 * ox + dx)) * C + c];
}

// ---------------------------------------------------------------------------
// Host orchestration
// ---------------------------------------------------------------------------
static void launch_gemm(int mode, const float* A, const float* W, const float* bias,
                        const float* res, float* out, int M, int N, int K, hipStream_t s)
{
    dim3 g((N + 63) / 64, (M + 63) / 64);
    if (mode == 0)      gemm_wmma<0><<<g, 128, 0, s>>>(A, W, bias, res, out, M, N, K);
    else if (mode == 1) gemm_wmma<1><<<g, 128, 0, s>>>(A, W, bias, res, out, M, N, K);
    else                gemm_wmma<2><<<g, 128, 0, s>>>(A, W, bias, res, out, M, N, K);
}

extern "C" void kernel_launch(void* const* d_in, const int* in_sizes, int n_in,
                              void* d_out, int out_size, void* d_ws, size_t ws_size,
                              hipStream_t stream)
{
    (void)in_sizes; (void)n_in; (void)out_size; (void)ws_size;
    const int Bq = 8;
    static const int EMBED[4]  = {128, 192, 256, 320};
    static const int DEPTHS[4] = {2, 2, 6, 2};
    static const int HEADS[4]  = {4, 6, 8, 10};

    int pi = 0;
    auto nx = [&]() { return (const float*)d_in[pi++]; };
    const float* X      = nx();
    const float* pe_w   = nx();
    const float* pe_b   = nx();
    const float* pe_nw  = nx();
    const float* pe_nb  = nx();
    const float* nw_    = nx();
    const float* nb_    = nx();
    const float* head_w = nx();
    const float* head_b = nx();

    const float* mg_nw[4] = {}; const float* mg_nb[4] = {}; const float* mg_w[4] = {};
    const float* bp[4][6][13];
    for (int i = 0; i < 4; ++i) {
        if (i > 0) { mg_nw[i] = nx(); mg_nb[i] = nx(); mg_w[i] = nx(); }
        for (int d = 0; d < DEPTHS[i]; ++d)
            for (int k = 0; k < 13; ++k) bp[i][d][k] = nx();
    }

    // Workspace carve (floats): x | t0 | t2 (each <= B*16384*128) | t1 (<= B*16384*512)
    const size_t XCAP = (size_t)Bq * 16384 * 128;  // 16.8M
    float* xb = (float*)d_ws;
    float* t0 = xb + XCAP;
    float* t2 = t0 + XCAP;
    float* t1 = t2 + XCAP;

    // Patch embed + LN
    {
        size_t total = (size_t)Bq * 128 * 128 * 128;
        patch_embed<<<dim3((unsigned)((total + 255) / 256)), 256, 0, stream>>>(X, pe_w, pe_b, t0, Bq);
        ln_kernel<<<dim3((unsigned)(Bq * 16384)), 128, 0, stream>>>(t0, xb, pe_nw, pe_nb, 128);
    }

    for (int i = 0; i < 4; ++i) {
        int Hres = 128 >> i;
        int C    = EMBED[i];
        int nh   = HEADS[i];
        int rows = Bq * Hres * Hres;

        if (i > 0) {
            int Hin = 256 >> i;            // = 2*Hres
            int Cin = EMBED[i - 1];
            size_t total = (size_t)Bq * Hres * Hres * 4 * Cin;
            merge_gather<<<dim3((unsigned)((total + 255) / 256)), 256, 0, stream>>>(xb, t0, Bq, Hin, Cin);
            ln_kernel<<<dim3((unsigned)rows), 128, 0, stream>>>(t0, t2, mg_nw[i], mg_nb[i], 4 * Cin);
            launch_gemm(0, t2, mg_w[i], nullptr, nullptr, xb, rows, C, 4 * Cin, stream);
        }

        int nWs    = Hres >> 3;
        int numWin = Bq * nWs * nWs;
        size_t etot = (size_t)rows * C;
        unsigned eblk = (unsigned)((etot + 255) / 256);

        for (int d = 0; d < DEPTHS[i]; ++d) {
            const float* const* P = bp[i][d];
            int shift = (d & 1) ? 4 : 0;

            // attention branch
            ln_kernel<<<dim3((unsigned)rows), 128, 0, stream>>>(xb, t2, P[0], P[1], C);
            win_part<<<dim3(eblk), 256, 0, stream>>>(t2, t0, Bq, Hres, C, shift);
            launch_gemm(0, t0, P[2], P[3], nullptr, t1, rows, 3 * C, C, stream);      // qkv
            win_attn<<<dim3((unsigned)numWin, (unsigned)nh), 128, 0, stream>>>(
                t1, t2, P[4], C, nh, nWs, Hres, shift);
            launch_gemm(0, t2, P[5], P[6], nullptr, t0, rows, C, C, stream);          // proj
            win_rev_add<<<dim3(eblk), 256, 0, stream>>>(t0, xb, Bq, Hres, C, shift);  // +residual

            // MLP branch
            ln_kernel<<<dim3((unsigned)rows), 128, 0, stream>>>(xb, t2, P[7], P[8], C);
            launch_gemm(1, t2, P[9],  P[10], nullptr, t1, rows, 4 * C, C, stream);    // fc1+GELU
            launch_gemm(2, t1, P[11], P[12], xb, xb, rows, C, 4 * C, stream);         // fc2+res
        }
    }

    // Final LN + head
    {
        int rows = Bq * 256;
        ln_kernel<<<dim3((unsigned)rows), 128, 0, stream>>>(xb, t0, nw_, nb_, 320);
        launch_gemm(0, t0, head_w, head_b, nullptr, (float*)d_out, rows, 32, 320, stream);
    }
}
